// LearnedCrossModalAttention_65094524338782
// MI455X (gfx1250) — compile-verified, matrix operand order checked by reading, not run on hardware
//
#include <hip/hip_runtime.h>

// B=65536, S=3, D=384, H=8, HD=48, L=3
// Batch-shared weights => every projection is a big GEMM on v_wmma_f32_16x16x32_bf16.
// One block = 16 batch rows, 256 threads (8 wave32).
// K/V projections: one job per n-tile with 3 independent accumulators (3 m-tiles
// share the B fragment) -> 3x less weight traffic + deep WMMA pipelining.

typedef __attribute__((ext_vector_type(16))) __bf16 v16bf;
typedef __attribute__((ext_vector_type(8)))  float  v8f;

#define D_DIM   384
#define S_KV    3
#define NHEAD   8
#define HDIM    48
#define NLAYER  3
#define MTILE   16
#define LDB     392   // bf16 row stride (padded: 784B -> 4-bank rotation per row)
#define LDF     388   // f32 row stride  (padded)

static __device__ __forceinline__ v16bf make_frag(const __bf16* p_lo, const __bf16* p_hi) {
    union { uint4 q[2]; v16bf v; } u;
    u.q[0] = *reinterpret_cast<const uint4*>(p_lo);   // elements 0..7  (k: base+8*hi .. +7)
    u.q[1] = *reinterpret_cast<const uint4*>(p_hi);   // elements 8..15 (k: base+16+8*hi .. +7)
    return u.v;
}

#define WMMA_BF16(a, b, c) \
    __builtin_amdgcn_wmma_f32_16x16x32_bf16(false, (a), false, (b), (short)0, (c), false, false)

// Single-tile K=384 GEMM with 1-deep fragment prefetch.
static __device__ __forceinline__ v8f gemm_k384(const __bf16* Arow, const __bf16* Brow,
                                                v8f acc, int hi) {
    v16bf a = make_frag(Arow + 8 * hi, Arow + 16 + 8 * hi);
    v16bf b = make_frag(Brow + 8 * hi, Brow + 16 + 8 * hi);
#pragma unroll
    for (int ks = 0; ks < 12; ++ks) {
        v16bf an, bn;
        if (ks < 11) {
            const int kb = (ks + 1) * 32;
            an = make_frag(Arow + kb + 8 * hi, Arow + kb + 16 + 8 * hi);
            bn = make_frag(Brow + kb + 8 * hi, Brow + kb + 16 + 8 * hi);
        }
        acc = WMMA_BF16(a, b, acc);
        if (ks < 11) { a = an; b = bn; }
    }
    return acc;
}

// Triple-tile K=384 GEMM: 3 A rows (m-tiles) share one B fragment stream.
static __device__ __forceinline__ void gemm_k384_x3(const __bf16* A0, const __bf16* A1,
                                                    const __bf16* A2, const __bf16* Brow,
                                                    v8f& c0, v8f& c1, v8f& c2, int hi) {
    v16bf b  = make_frag(Brow + 8 * hi, Brow + 16 + 8 * hi);
    v16bf a0 = make_frag(A0 + 8 * hi, A0 + 16 + 8 * hi);
    v16bf a1 = make_frag(A1 + 8 * hi, A1 + 16 + 8 * hi);
    v16bf a2 = make_frag(A2 + 8 * hi, A2 + 16 + 8 * hi);
#pragma unroll
    for (int ks = 0; ks < 12; ++ks) {
        v16bf bn, a0n, a1n, a2n;
        if (ks < 11) {
            const int kb = (ks + 1) * 32;
            bn  = make_frag(Brow + kb + 8 * hi, Brow + kb + 16 + 8 * hi);
            a0n = make_frag(A0 + kb + 8 * hi, A0 + kb + 16 + 8 * hi);
            a1n = make_frag(A1 + kb + 8 * hi, A1 + kb + 16 + 8 * hi);
            a2n = make_frag(A2 + kb + 8 * hi, A2 + kb + 16 + 8 * hi);
        }
        c0 = WMMA_BF16(a0, b, c0);
        c1 = WMMA_BF16(a1, b, c1);
        c2 = WMMA_BF16(a2, b, c2);
        if (ks < 11) { b = bn; a0 = a0n; a1 = a1n; a2 = a2n; }
    }
}

__global__ __launch_bounds__(256)
void convert_bf16_kernel(const float* __restrict__ src, unsigned short* __restrict__ dst, int n) {
    int i = blockIdx.x * 256 + threadIdx.x;
    if (i < n) {
        __bf16 b = (__bf16)src[i];
        reinterpret_cast<__bf16*>(dst)[i] = b;
    }
}

__global__ __launch_bounds__(256)
void cross_attn_stack_kernel(const float* __restrict__ q,
                             const float* __restrict__ kv,
                             const unsigned short* __restrict__ inW_u,
                             const float* __restrict__ inB,
                             const unsigned short* __restrict__ outW_u,
                             const float* __restrict__ outB,
                             const float* __restrict__ lnW,
                             const float* __restrict__ lnB,
                             const unsigned short* __restrict__ projW_u,
                             const float* __restrict__ projB,
                             float* __restrict__ out) {
    const __bf16* inWb   = reinterpret_cast<const __bf16*>(inW_u);
    const __bf16* outWb  = reinterpret_cast<const __bf16*>(outW_u);
    const __bf16* projWb = reinterpret_cast<const __bf16*>(projW_u);

    __shared__ __align__(16) float  s_out_f[MTILE * LDF];        // residual stream (f32)
    __shared__ __align__(16) __bf16 s_out_b[MTILE * LDB];        // bf16 A-copy / ctx buffer
    __shared__ __align__(16) __bf16 s_kv  [MTILE * S_KV * LDB];  // kv rows (bf16)
    __shared__ __align__(16) __bf16 s_qp  [MTILE * LDB];
    __shared__ __align__(16) __bf16 s_kp  [MTILE * S_KV * LDB];
    __shared__ __align__(16) __bf16 s_vp  [MTILE * S_KV * LDB];

    const int b0   = blockIdx.x * MTILE;
    const int tid  = threadIdx.x;
    const int lane = tid & 31;
    const int wave = tid >> 5;
    const int nloc = lane & 15;
    const int hi   = lane >> 4;

    // ---- Stage inputs ----
    for (int idx = tid; idx < MTILE * D_DIM; idx += 256) {
        int m = idx / D_DIM, d = idx - m * D_DIM;
        float v = q[(size_t)(b0 + m) * D_DIM + d];
        s_out_f[m * LDF + d] = v;
        s_out_b[m * LDB + d] = (__bf16)v;
    }
    for (int idx = tid; idx < MTILE * S_KV * D_DIM; idx += 256) {
        int r = idx / D_DIM, d = idx - r * D_DIM;
        s_kv[r * LDB + d] = (__bf16)kv[(size_t)b0 * S_KV * D_DIM + idx];
    }
    __syncthreads();

    for (int l = 0; l < NLAYER; ++l) {
        const __bf16* Wl = inWb + (size_t)l * (3 * D_DIM) * D_DIM;
        const float*  Bl = inB  + l * (3 * D_DIM);

        // ---- Q/K/V projections: 72 jobs (24 Q single-acc, 24 K + 24 V triple-acc) ----
        for (int t = 0; t < 9; ++t) {
            const int j    = t * 8 + wave;      // interleave: each wave gets 3 Q, 3 K, 3 V
            const int part = j / 24;            // 0=Q, 1=K, 2=V
            const int nt   = j - part * 24;
            const int ng   = part * D_DIM + nt * 16 + nloc;
            const __bf16* Brow = Wl + (size_t)ng * D_DIM;
            const float bias = Bl[ng];
            const int ncol = nt * 16 + nloc;
            if (part == 0) {                    // Q projection: 16 rows
                v8f acc;
#pragma unroll
                for (int i = 0; i < 8; ++i) acc[i] = bias;
                acc = gemm_k384(s_out_b + nloc * LDB, Brow, acc, hi);
#pragma unroll
                for (int v = 0; v < 8; ++v)
                    s_qp[(v + 8 * hi) * LDB + ncol] = (__bf16)acc[v];
            } else {                            // K/V projection: 48 rows, shared B stream
                __bf16* dst = (part == 2) ? s_vp : s_kp;
                v8f c0, c1, c2;
#pragma unroll
                for (int i = 0; i < 8; ++i) { c0[i] = bias; c1[i] = bias; c2[i] = bias; }
                gemm_k384_x3(s_kv + (0  + nloc) * LDB,
                             s_kv + (16 + nloc) * LDB,
                             s_kv + (32 + nloc) * LDB,
                             Brow, c0, c1, c2, hi);
#pragma unroll
                for (int v = 0; v < 8; ++v) {
                    const int r = v + 8 * hi;
                    dst[(r     ) * LDB + ncol] = (__bf16)c0[v];
                    dst[(r + 16) * LDB + ncol] = (__bf16)c1[v];
                    dst[(r + 32) * LDB + ncol] = (__bf16)c2[v];
                }
            }
        }
        __syncthreads();

        // ---- Attention: 128 threads, one (row m, head h) each ----
        if (tid < MTILE * NHEAD) {
            const int m = tid >> 3, h = tid & 7;
            const __bf16* qrow = s_qp + m * LDB + h * HDIM;
            float sc[S_KV];
#pragma unroll
            for (int jk = 0; jk < S_KV; ++jk) {
                const __bf16* krow = s_kp + (m * S_KV + jk) * LDB + h * HDIM;
                float s = 0.f;
#pragma unroll
                for (int d = 0; d < HDIM; ++d) s += (float)qrow[d] * (float)krow[d];
                sc[jk] = s * 0.14433756729740643f;   // 1/sqrt(48)
            }
            float mx = fmaxf(sc[0], fmaxf(sc[1], sc[2]));
            float e0 = __expf(sc[0] - mx), e1 = __expf(sc[1] - mx), e2 = __expf(sc[2] - mx);
            float inv = 1.0f / (e0 + e1 + e2);
            e0 *= inv; e1 *= inv; e2 *= inv;
            const __bf16* v0 = s_vp + (m * S_KV + 0) * LDB + h * HDIM;
            const __bf16* v1 = s_vp + (m * S_KV + 1) * LDB + h * HDIM;
            const __bf16* v2 = s_vp + (m * S_KV + 2) * LDB + h * HDIM;
            __bf16* crow = s_out_b + m * LDB + h * HDIM;   // ctx reuses bf16 A-buffer
#pragma unroll
            for (int d = 0; d < HDIM; ++d) {
                float c = e0 * (float)v0[d] + e1 * (float)v1[d] + e2 * (float)v2[d];
                crow[d] = (__bf16)c;
            }
        }
        __syncthreads();

        // ---- out_proj: 24 tile jobs, accumulate (bias + GEMM) into f32 residual ----
        const __bf16* Wo = outWb + (size_t)l * D_DIM * D_DIM;
        const float*  Bo = outB  + l * D_DIM;
        for (int t = 0; t < 3; ++t) {
            const int nt = wave * 3 + t;
            const int ng = nt * 16 + nloc;
            const __bf16* Brow = Wo + (size_t)ng * D_DIM;
            float bias = Bo[ng];
            v8f acc;
#pragma unroll
            for (int i = 0; i < 8; ++i) acc[i] = bias;
            acc = gemm_k384(s_out_b + nloc * LDB, Brow, acc, hi);
#pragma unroll
            for (int v = 0; v < 8; ++v)
                s_out_f[(v + 8 * hi) * LDF + ng] += acc[v];
        }
        __syncthreads();

        // ---- LayerNorm: one wave per 2 rows, shfl_xor wave reduction ----
        for (int rr = 0; rr < 2; ++rr) {
            const int r = wave * 2 + rr;
            float s = 0.f;
#pragma unroll
            for (int i = 0; i < 12; ++i) s += s_out_f[r * LDF + lane + 32 * i];
#pragma unroll
            for (int off = 16; off; off >>= 1) s += __shfl_xor(s, off, 32);
            const float mean = s * (1.0f / 384.0f);
            float vs = 0.f;
#pragma unroll
            for (int i = 0; i < 12; ++i) {
                float d = s_out_f[r * LDF + lane + 32 * i] - mean;
                vs += d * d;
            }
#pragma unroll
            for (int off = 16; off; off >>= 1) vs += __shfl_xor(vs, off, 32);
            const float rstd = rsqrtf(vs * (1.0f / 384.0f) + 1e-5f);
#pragma unroll
            for (int i = 0; i < 12; ++i) {
                int c = lane + 32 * i;
                float x = s_out_f[r * LDF + c];
                float y = (x - mean) * rstd * lnW[l * D_DIM + c] + lnB[l * D_DIM + c];
                s_out_f[r * LDF + c] = y;
                s_out_b[r * LDB + c] = (__bf16)y;
            }
        }
        __syncthreads();
    }

    // ---- Final projection -> global f32 output ----
    for (int t = 0; t < 3; ++t) {
        const int nt = wave * 3 + t;
        const int ng = nt * 16 + nloc;
        const __bf16* Brow = projWb + (size_t)ng * D_DIM;
        float bias = projB[ng];
        v8f acc;
#pragma unroll
        for (int i = 0; i < 8; ++i) acc[i] = bias;
        acc = gemm_k384(s_out_b + nloc * LDB, Brow, acc, hi);
#pragma unroll
        for (int v = 0; v < 8; ++v)
            out[(size_t)(b0 + v + 8 * hi) * D_DIM + ng] = acc[v];
    }
}

extern "C" void kernel_launch(void* const* d_in, const int* in_sizes, int n_in,
                              void* d_out, int out_size, void* d_ws, size_t ws_size,
                              hipStream_t stream) {
    const float* q      = (const float*)d_in[0];
    const float* kv     = (const float*)d_in[1];
    const float* in_w   = (const float*)d_in[2];
    const float* in_b   = (const float*)d_in[3];
    const float* out_w  = (const float*)d_in[4];
    const float* out_b  = (const float*)d_in[5];
    const float* ln_w   = (const float*)d_in[6];
    const float* ln_b   = (const float*)d_in[7];
    const float* proj_w = (const float*)d_in[8];
    const float* proj_b = (const float*)d_in[9];
    float* out = (float*)d_out;

    unsigned short* wbf = (unsigned short*)d_ws;
    const int C_in  = 3 * 1152 * 384;   // in_proj_w elements
    const int C_out = 3 * 384 * 384;    // out_proj_w elements
    const int C_p   = 384 * 384;        // proj_w elements

    convert_bf16_kernel<<<(C_in  + 255) / 256, 256, 0, stream>>>(in_w,   wbf,                 C_in);
    convert_bf16_kernel<<<(C_out + 255) / 256, 256, 0, stream>>>(out_w,  wbf + C_in,          C_out);
    convert_bf16_kernel<<<(C_p   + 255) / 256, 256, 0, stream>>>(proj_w, wbf + C_in + C_out,  C_p);

    const int blocks = 65536 / MTILE;   // 4096
    cross_attn_stack_kernel<<<blocks, 256, 0, stream>>>(
        q, kv,
        wbf, in_b,
        wbf + C_in, out_b,
        ln_w, ln_b,
        wbf + C_in + C_out, proj_b,
        out);
}
